// NeighborSamplingGCN_77584289235240
// MI455X (gfx1250) — compile-verified, matrix operand order checked by reading, not run on hardware
//
#include <hip/hip_runtime.h>
#include <hip/hip_bf16.h>

// ---------------------------------------------------------------------------
// Binarized 2-layer GraphSAGE for MI455X (gfx1250, wave32, WMMA).
// norm_bin forward = sign(x - colmean): std is dead code -> only col-means.
// GEMMs: v_wmma_f32_16x16x32_f16, NT tiles per wave (A-fragment reuse),
// weights pre-packed into B-fragment layout (2x b128 loads per fragment).
// Scatter-aggregation uses exact (integer-valued) f32 atomics -> deterministic.
// ---------------------------------------------------------------------------

typedef __attribute__((ext_vector_type(16))) _Float16 v16h;
typedef __attribute__((ext_vector_type(8)))  float    v8f;

#define N0 400000
#define N1 45056
#define N2 4096
#define E0 1126400
#define E1 102400
#define D_IN 256
#define D_HID 256
#define D_OUT 48
#define STAT_BLOCKS 512

// ---------------- column means (deterministic two-pass) --------------------
__global__ void colmean_partial(const float* __restrict__ src, int rows,
                                float* __restrict__ partial) {
  int t = threadIdx.x;              // feature column 0..255
  int b = blockIdx.x;
  int nb = gridDim.x;
  float s = 0.f;
  for (int r = b; r < rows; r += nb)          // block reads full rows: coalesced
    s += src[(size_t)r * 256 + t];
  partial[(size_t)b * 256 + t] = s;
}

__global__ void colmean_final(const float* __restrict__ partial, int nblk,
                              int rows, float* __restrict__ mean) {
  int t = threadIdx.x;
  float s = 0.f;
  for (int b = 0; b < nblk; ++b) s += partial[(size_t)b * 256 + t];
  mean[t] = s / (float)rows;
}

// ---------------- edge scatter of sign(x - mean) ---------------------------
__global__ void scatter_sign(const float* __restrict__ feat,
                             const float* __restrict__ mean,
                             const int* __restrict__ src,
                             const int* __restrict__ dst,
                             float* __restrict__ agg,
                             float* __restrict__ cnt,
                             int nEdges, int edgesPerBlock) {
  int t = threadIdx.x;              // feature column 0..255
  float m = mean[t];
  int e0 = blockIdx.x * edgesPerBlock;
  int e1 = e0 + edgesPerBlock; if (e1 > nEdges) e1 = nEdges;
  for (int e = e0; e < e1; ++e) {
    int s = src[e];
    int d = dst[e];
    float v = feat[(size_t)s * 256 + t] - m;
    float sg = (v > 0.f) ? 1.f : ((v < 0.f) ? -1.f : 0.f);
    atomicAdd(&agg[(size_t)d * 256 + t], sg);        // exact: integer-valued
    if (t == 0) atomicAdd(&cnt[d], 1.f);
  }
}

// ---------------- weight pre-pack into WMMA B-fragment layout --------------
// B 32x16 f16 (ISA 7.12.2): lane<16 -> N=lane,    K = {k0+0..7, k0+16..23}
//                           lane>=16 -> N=lane-16, K = {k0+8..15, k0+24..31}
// Packed: frag(tn, kstep, lane) = out + (((tn*8)+kstep)*32 + lane)*16 halves
// -> each fragment is one contiguous, 32B-aligned v16h.
__global__ void pack_b(const float* __restrict__ W, int ldb, int tilesN,
                       _Float16* __restrict__ out) {
  int idx = blockIdx.x * blockDim.x + threadIdx.x;
  int total = tilesN * 8 * 32;
  if (idx >= total) return;
  int lane  = idx & 31;
  int kstep = (idx >> 5) & 7;
  int tn    = idx >> 8;
  int kA = kstep * 32 + ((lane & 16) ? 8 : 0);
  int n  = tn * 16 + (lane & 15);
  _Float16* o = out + (size_t)idx * 16;
#pragma unroll
  for (int i = 0; i < 8; ++i) {
    o[i]     = (_Float16)W[(size_t)(kA + i) * ldb + n];
    o[8 + i] = (_Float16)W[(size_t)(kA + 16 + i) * ldb + n];
  }
}

// ---------------- WMMA A-fragment loaders (ISA 7.12.2 16-bit layout) -------
// A 16x32 f16: lane<16 -> M=lane,    K = {k0+0..7, k0+16..23}
//              lane>=16 -> M=lane-16, K = {k0+8..15, k0+24..31}
__device__ __forceinline__ v16h load_a_scaled(const float* __restrict__ row,
                                              int k0, int lane, float scale) {
  int kA = k0 + ((lane & 16) ? 8 : 0);
  const float4 f0 = *reinterpret_cast<const float4*>(row + kA);
  const float4 f1 = *reinterpret_cast<const float4*>(row + kA + 4);
  const float4 f2 = *reinterpret_cast<const float4*>(row + kA + 16);
  const float4 f3 = *reinterpret_cast<const float4*>(row + kA + 20);
  float t[16] = {f0.x, f0.y, f0.z, f0.w, f1.x, f1.y, f1.z, f1.w,
                 f2.x, f2.y, f2.z, f2.w, f3.x, f3.y, f3.z, f3.w};
  v16h r;
#pragma unroll
  for (int i = 0; i < 16; ++i) r[i] = (_Float16)(t[i] * scale);
  return r;
}

__device__ __forceinline__ float sgnf(float v) {
  return (v > 0.f) ? 1.f : ((v < 0.f) ? -1.f : 0.f);
}

__device__ __forceinline__ v16h load_a_sign(const float* __restrict__ row,
                                            const float* __restrict__ mean,
                                            int k0, int lane) {
  int kA = k0 + ((lane & 16) ? 8 : 0);
  const float4 f0 = *reinterpret_cast<const float4*>(row + kA);
  const float4 f1 = *reinterpret_cast<const float4*>(row + kA + 4);
  const float4 f2 = *reinterpret_cast<const float4*>(row + kA + 16);
  const float4 f3 = *reinterpret_cast<const float4*>(row + kA + 20);
  const float4 m0 = *reinterpret_cast<const float4*>(mean + kA);
  const float4 m1 = *reinterpret_cast<const float4*>(mean + kA + 4);
  const float4 m2 = *reinterpret_cast<const float4*>(mean + kA + 16);
  const float4 m3 = *reinterpret_cast<const float4*>(mean + kA + 20);
  float t[16] = {f0.x - m0.x, f0.y - m0.y, f0.z - m0.z, f0.w - m0.w,
                 f1.x - m1.x, f1.y - m1.y, f1.z - m1.z, f1.w - m1.w,
                 f2.x - m2.x, f2.y - m2.y, f2.z - m2.z, f2.w - m2.w,
                 f3.x - m3.x, f3.y - m3.y, f3.z - m3.z, f3.w - m3.w};
  v16h r;
#pragma unroll
  for (int i = 0; i < 16; ++i) r[i] = (_Float16)sgnf(t[i]);
  return r;
}

// ---------------- dual-input WMMA GEMM, NT tiles per wave ------------------
// out[m,n] = relu?( (agg[m,:]/max(cnt[m],1)) @ Wl + sign(feat[m,:]-mean_t) @ Wr + bias )
// One wave computes a 16 x (NT*16) strip: A fragments reused NT times.
// Inner WMMAs sweep distinct accumulators -> no D->A RAW hazard stalls.
template <int NT>
__global__ void gemm_dual_wmma(const float* __restrict__ agg,
                               const float* __restrict__ cnt,
                               const float* __restrict__ feat,
                               const float* __restrict__ mean_t,
                               const _Float16* __restrict__ Blpk,
                               const _Float16* __restrict__ Brpk,
                               const float* __restrict__ bias,
                               float* __restrict__ outp,
                               int nGroups, int ldc, int doRelu) {
  int lane = threadIdx.x & 31;
  int wave = threadIdx.x >> 5;
  int task = blockIdx.x * (blockDim.x >> 5) + wave;
  int tm     = task / nGroups;
  int tnBase = (task % nGroups) * NT;

  int m = tm * 16 + (lane & 15);        // A-fragment row for this lane
  const float* aggRow = agg + (size_t)m * 256;
  const float* xRow   = feat + (size_t)m * 256;
  float scale = 1.f / fmaxf(cnt[m], 1.f);

  v8f acc[NT];
#pragma unroll
  for (int j = 0; j < NT; ++j) acc[j] = (v8f){};

#pragma unroll
  for (int kstep = 0; kstep < 8; ++kstep) {
    int k0 = kstep * 32;
    v16h a1 = load_a_scaled(aggRow, k0, lane, scale);
    v16h a2 = load_a_sign(xRow, mean_t, k0, lane);
    const _Float16* fbase =
        (const _Float16*)nullptr;  // (avoid unused warnings on some compilers)
    (void)fbase;
#pragma unroll
    for (int j = 0; j < NT; ++j) {
      size_t fi = (((size_t)(tnBase + j) * 8 + kstep) * 32 + lane) * 16;
      v16h b1 = *reinterpret_cast<const v16h*>(Blpk + fi);
      acc[j] = __builtin_amdgcn_wmma_f32_16x16x32_f16(false, a1, false, b1,
                                                      (short)0, acc[j], false, false);
    }
#pragma unroll
    for (int j = 0; j < NT; ++j) {
      size_t fi = (((size_t)(tnBase + j) * 8 + kstep) * 32 + lane) * 16;
      v16h b2 = *reinterpret_cast<const v16h*>(Brpk + fi);
      acc[j] = __builtin_amdgcn_wmma_f32_16x16x32_f16(false, a2, false, b2,
                                                      (short)0, acc[j], false, false);
    }
  }

  // D layout: VGPR v, lane l -> (M = tm*16 + v + (l>=16 ? 8 : 0), N = tn*16 + l%16)
  int mb = tm * 16 + ((lane & 16) ? 8 : 0);
#pragma unroll
  for (int j = 0; j < NT; ++j) {
    int n = (tnBase + j) * 16 + (lane & 15);
    float bv = bias[n];
#pragma unroll
    for (int v = 0; v < 8; ++v) {
      float val = acc[j][v] + bv;
      if (doRelu) val = fmaxf(val, 0.f);
      outp[(size_t)(mb + v) * ldc + n] = val;
    }
  }
}

// ---------------- row-wise log_softmax over 48 classes ---------------------
__global__ void logsoftmax48(const float* __restrict__ pre,
                             float* __restrict__ out, int rows) {
  int r = blockIdx.x * blockDim.x + threadIdx.x;
  if (r >= rows) return;
  const float* p = pre + (size_t)r * 48;
  float mx = p[0];
#pragma unroll
  for (int j = 1; j < 48; ++j) mx = fmaxf(mx, p[j]);
  float s = 0.f;
#pragma unroll
  for (int j = 0; j < 48; ++j) s += __expf(p[j] - mx);
  float lse = mx + __logf(s);
  float* o = out + (size_t)r * 48;
#pragma unroll
  for (int j = 0; j < 48; ++j) o[j] = p[j] - lse;
}

// ---------------------------------------------------------------------------
extern "C" void kernel_launch(void* const* d_in, const int* in_sizes, int n_in,
                              void* d_out, int out_size, void* d_ws, size_t ws_size,
                              hipStream_t stream) {
  const float* x    = (const float*)d_in[0];
  const int*   src0 = (const int*)d_in[1];
  const int*   dst0 = (const int*)d_in[2];
  const int*   src1 = (const int*)d_in[3];
  const int*   dst1 = (const int*)d_in[4];
  const float* W1l  = (const float*)d_in[5];
  const float* b1   = (const float*)d_in[6];
  const float* W1r  = (const float*)d_in[7];
  const float* W2l  = (const float*)d_in[8];
  const float* b2   = (const float*)d_in[9];
  const float* W2r  = (const float*)d_in[10];
  float* out = (float*)d_out;

  // workspace carve-out (256B aligned)
  char* ws = (char*)d_ws;
  size_t off = 0;
  auto carve = [&](size_t bytes) {
    void* p = ws + off;
    off += (bytes + 255) & ~(size_t)255;
    return p;
  };
  float*     P       = (float*)carve((size_t)STAT_BLOCKS * 256 * 4);
  float*     mean_xb = (float*)carve(256 * 4);
  float*     mean_xt = (float*)carve(256 * 4);
  float*     mean_hb = (float*)carve(256 * 4);
  float*     mean_ht = (float*)carve(256 * 4);
  float*     agg1    = (float*)carve((size_t)N1 * 256 * 4);
  float*     cnt1    = (float*)carve((size_t)N1 * 4);
  float*     h       = (float*)carve((size_t)N1 * 256 * 4);
  float*     agg2    = (float*)carve((size_t)N2 * 256 * 4);
  float*     cnt2    = (float*)carve((size_t)N2 * 4);
  float*     pre     = (float*)carve((size_t)N2 * 48 * 4);
  _Float16*  B1l_pk  = (_Float16*)carve((size_t)16 * 8 * 32 * 16 * 2);
  _Float16*  B1r_pk  = (_Float16*)carve((size_t)16 * 8 * 32 * 16 * 2);
  _Float16*  B2l_pk  = (_Float16*)carve((size_t)3 * 8 * 32 * 16 * 2);
  _Float16*  B2r_pk  = (_Float16*)carve((size_t)3 * 8 * 32 * 16 * 2);

  // ---- pre-pack weights into WMMA B-fragment layout (f16) ----
  pack_b<<<16, 256, 0, stream>>>(W1l, D_HID, 16, B1l_pk);
  pack_b<<<16, 256, 0, stream>>>(W1r, D_HID, 16, B1r_pk);
  pack_b<<<3, 256, 0, stream>>>(W2l, D_OUT, 3, B2l_pk);
  pack_b<<<3, 256, 0, stream>>>(W2r, D_OUT, 3, B2r_pk);

  // ---- layer 0 column means (over all N0 rows and target slice N1) ----
  colmean_partial<<<STAT_BLOCKS, 256, 0, stream>>>(x, N0, P);
  colmean_final<<<1, 256, 0, stream>>>(P, STAT_BLOCKS, N0, mean_xb);
  colmean_partial<<<STAT_BLOCKS, 256, 0, stream>>>(x, N1, P);
  colmean_final<<<1, 256, 0, stream>>>(P, STAT_BLOCKS, N1, mean_xt);

  // ---- layer 0 aggregation (sign scatter, exact f32 atomics) ----
  hipMemsetAsync(agg1, 0, (size_t)N1 * 256 * 4, stream);
  hipMemsetAsync(cnt1, 0, (size_t)N1 * 4, stream);
  {
    const int epb = 16;
    scatter_sign<<<(E0 + epb - 1) / epb, 256, 0, stream>>>(
        x, mean_xb, src0, dst0, agg1, cnt1, E0, epb);
  }

  // ---- GEMM1: h = relu(mean(agg1) @ W1l + sign(x[:N1]-mean_xt) @ W1r + b1)
  // NT=8 -> 2 N-groups, tasks = 2816*2 = 5632, 8 waves/block -> 704 blocks.
  gemm_dual_wmma<8><<<704, 256, 0, stream>>>(
      agg1, cnt1, x, mean_xt, B1l_pk, B1r_pk, b1, h, /*nGroups=*/2,
      /*ldc=*/D_HID, /*doRelu=*/1);

  // ---- layer 1 column means over h ----
  colmean_partial<<<STAT_BLOCKS, 256, 0, stream>>>(h, N1, P);
  colmean_final<<<1, 256, 0, stream>>>(P, STAT_BLOCKS, N1, mean_hb);
  colmean_partial<<<STAT_BLOCKS, 256, 0, stream>>>(h, N2, P);
  colmean_final<<<1, 256, 0, stream>>>(P, STAT_BLOCKS, N2, mean_ht);

  // ---- layer 1 aggregation ----
  hipMemsetAsync(agg2, 0, (size_t)N2 * 256 * 4, stream);
  hipMemsetAsync(cnt2, 0, (size_t)N2 * 4, stream);
  {
    const int epb = 16;
    scatter_sign<<<(E1 + epb - 1) / epb, 256, 0, stream>>>(
        h, mean_hb, src1, dst1, agg2, cnt2, E1, epb);
  }

  // ---- GEMM2: pre = mean(agg2) @ W2l + sign(h[:N2]-mean_ht) @ W2r + b2 ----
  // NT=3 -> 1 N-group, tasks = 256, 8 waves/block -> 32 blocks.
  gemm_dual_wmma<3><<<32, 256, 0, stream>>>(
      agg2, cnt2, h, mean_ht, B2l_pk, B2r_pk, b2, pre, /*nGroups=*/1,
      /*ldc=*/D_OUT, /*doRelu=*/0);

  // ---- log_softmax over 48 classes ----
  logsoftmax48<<<(N2 + 255) / 256, 256, 0, stream>>>(pre, out, N2);
}
